// SpectralMambaBlock_35768487641359
// MI455X (gfx1250) — matrix operand order repaired and checked by reading, not compiled, not run on hardware
//
#include <hip/hip_runtime.h>
#include <hip/hip_bf16.h>

typedef _Float16 v8h  __attribute__((ext_vector_type(8)));
typedef _Float16 v16h __attribute__((ext_vector_type(16)));
typedef float    v8f  __attribute__((ext_vector_type(8)));

#define D_MODEL 256
#define D_STATE 16
#define D_INNER 512
#define DT_RANK 16
#define NB 16
#define LL 2048
#define BL (NB * LL)   // 32768 token rows

union F16Frag { v16h v; v8h h[2]; };

__device__ __forceinline__ float sigmoidf_(float x) { return 1.0f / (1.0f + __expf(-x)); }

// ---------------------------------------------------------------------------
// Tensor Data Mover support (probe-verified builtin; arity differs by clang)
// ---------------------------------------------------------------------------
#if defined(__has_builtin)
# if __has_builtin(__builtin_amdgcn_tensor_load_to_lds) && __has_builtin(__builtin_amdgcn_s_wait_tensorcnt)
#  define USE_TDM 1
# endif
#endif
#ifndef USE_TDM
# define USE_TDM 0
#endif

#if USE_TDM
typedef unsigned int uv4 __attribute__((ext_vector_type(4)));
typedef int          iv4 __attribute__((ext_vector_type(4)));
typedef int          iv8 __attribute__((ext_vector_type(8)));

// Flat 1-D tile copy: 3072 f32 (12 KB) global -> LDS via TDM descriptor.
__device__ __forceinline__ void tdm_copy_12k(const float* gsrc, unsigned lds_addr) {
    unsigned long long ga = (unsigned long long)(size_t)gsrc;
    uv4 g0;
    g0[0] = 1u;                                      // count=1 (valid user D#)
    g0[1] = lds_addr;                                // lds_addr (bytes)
    g0[2] = (unsigned)(ga & 0xFFFFFFFFu);            // global_addr[31:0]
    g0[3] = (unsigned)((ga >> 32) & 0x01FFFFFFu)     // global_addr[56:32]
          | (2u << 30);                              // type = 2
    iv8 g1 = (iv8)0;
    g1[0] = (int)(2u << 16);                         // data_size = 4 bytes
    g1[1] = (int)(3072u << 16);                      // tensor_dim0[15:0]
    g1[2] = (int)(1u << 16);                         // tensor_dim0 hi=0, tensor_dim1=1
    g1[3] = (int)(3072u << 16);                      // tile_dim0 = 3072
    g1[5] = 3072;                                    // tensor_dim0_stride
    iv4 gz = (iv4)0;
#if __clang_major__ >= 23
    iv8 gz8 = (iv8)0;
    __builtin_amdgcn_tensor_load_to_lds(g0, g1, gz, gz, gz8, 0);
#else
    __builtin_amdgcn_tensor_load_to_lds(g0, g1, gz, gz, 0);
#endif
}
#endif

// ---------------------------------------------------------------------------
// LayerNorm: one wave per 256-elem row, 8 rows per 256-thread block. f32 -> f16
// ---------------------------------------------------------------------------
__global__ void __launch_bounds__(256) ln_kernel(const float* __restrict__ x,
                                                 const float* __restrict__ w,
                                                 const float* __restrict__ b,
                                                 _Float16* __restrict__ xn) {
    int row  = blockIdx.x * 8 + (threadIdx.x >> 5);
    int lane = threadIdx.x & 31;
    const float* xr = x + (size_t)row * D_MODEL;
    float4 v0 = ((const float4*)xr)[lane * 2 + 0];
    float4 v1 = ((const float4*)xr)[lane * 2 + 1];
    float s = v0.x + v0.y + v0.z + v0.w + v1.x + v1.y + v1.z + v1.w;
    float q = v0.x*v0.x + v0.y*v0.y + v0.z*v0.z + v0.w*v0.w
            + v1.x*v1.x + v1.y*v1.y + v1.z*v1.z + v1.w*v1.w;
    #pragma unroll
    for (int off = 16; off; off >>= 1) {
        s += __shfl_xor(s, off, 32);
        q += __shfl_xor(q, off, 32);
    }
    float mu  = s * (1.0f / D_MODEL);
    float var = q * (1.0f / D_MODEL) - mu * mu;
    float rs  = rsqrtf(var + 1e-5f);
    int k = lane * 8;
    float in[8] = {v0.x, v0.y, v0.z, v0.w, v1.x, v1.y, v1.z, v1.w};
    v8h o;
    #pragma unroll
    for (int i = 0; i < 8; i++)
        o[i] = (_Float16)((in[i] - mu) * rs * w[k + i] + b[k + i]);
    ((v8h*)(xn + (size_t)row * D_MODEL))[lane] = o;
}

// f32 -> f16 weight conversion
__global__ void cvt_f32_f16(const float* __restrict__ s, _Float16* __restrict__ d, int n) {
    int i = blockIdx.x * blockDim.x + threadIdx.x;
    if (i < n) d[i] = (_Float16)s[i];
}

// ---------------------------------------------------------------------------
// WMMA fragment loaders (contiguous k = 16h + s order in A and B, wave32)
// ---------------------------------------------------------------------------
__device__ __forceinline__ void load_a_frag(F16Frag& f, const _Float16* __restrict__ A,
                                            int lda, int mBase, int kk, int lane) {
    int hf = lane >> 4, mr = lane & 15;
    const _Float16* p = A + (size_t)(mBase + mr) * lda + kk + hf * 16;
    f.h[0] = *(const v8h*)(p);        // K = kk + hf*16 .. +7
    f.h[1] = *(const v8h*)(p + 8);    // K = kk + hf*16 + 8 .. +15
}

// B[k][n] = W[n][k] (W stored (N,K) row-major): contiguous 16-half chunk per lane
__device__ __forceinline__ v16h load_b_frag(const _Float16* __restrict__ W,
                                            int ldw, int nBase, int kk, int lane) {
    int hf = lane >> 4, nr = lane & 15;
    const _Float16* p = W + (size_t)(nBase + nr) * ldw + kk + hf * 16;
    return *(const v16h*)(p);
}

__device__ __forceinline__ void store_tile_f16(const v8f& c, _Float16* __restrict__ dst,
                                               int ldd, int mBase, int n, int lane) {
    int m0 = mBase + ((lane >> 4) << 3);
    int nc = n + (lane & 15);
    #pragma unroll
    for (int r = 0; r < 8; r++) dst[(size_t)(m0 + r) * ldd + nc] = (_Float16)c[r];
}

__device__ __forceinline__ void store_tile_f32(const v8f& c, float* __restrict__ dst,
                                               int ldd, int mBase, int n, int lane) {
    int m0 = mBase + ((lane >> 4) << 3);
    int nc = n + (lane & 15);
    #pragma unroll
    for (int r = 0; r < 8; r++) dst[(size_t)(m0 + r) * ldd + nc] = c[r];
}

#define WMMA_F16(A_, B_, C_) \
    __builtin_amdgcn_wmma_f32_16x16x32_f16(false, (A_), false, (B_), (short)0, (C_), false, false)

// ---------------------------------------------------------------------------
// GEMM1: xz = xn(32768x256) * in_proj_w^T(256x1024); split -> x_in, z (f16)
// grid (256,16), block 128 (4 waves); wave = 32x64 tile, double-buffered K
// ---------------------------------------------------------------------------
__global__ void __launch_bounds__(128) gemm_in(const _Float16* __restrict__ A,
                                               const _Float16* __restrict__ W,
                                               _Float16* __restrict__ xin,
                                               _Float16* __restrict__ zz) {
    const int K = 256;
    int lane  = threadIdx.x & 31;
    int wv    = threadIdx.x >> 5;
    int mBase = blockIdx.x * 128 + wv * 32;
    int nBase = blockIdx.y * 64;
    v8f acc[2][4] = {};
    F16Frag a[2][2];
    v16h    b[2][4];
    load_a_frag(a[0][0], A, K, mBase,      0, lane);
    load_a_frag(a[0][1], A, K, mBase + 16, 0, lane);
    #pragma unroll
    for (int t = 0; t < 4; t++) b[0][t] = load_b_frag(W, K, nBase + t * 16, 0, lane);
    #pragma unroll
    for (int kk = 0; kk < K; kk += 32) {
        const int cb = (kk >> 5) & 1, nb = cb ^ 1;
        if (kk + 32 < K) {
            load_a_frag(a[nb][0], A, K, mBase,      kk + 32, lane);
            load_a_frag(a[nb][1], A, K, mBase + 16, kk + 32, lane);
            #pragma unroll
            for (int t = 0; t < 4; t++) b[nb][t] = load_b_frag(W, K, nBase + t * 16, kk + 32, lane);
        }
        if (kk + 64 < K)
            __builtin_prefetch(A + (size_t)(mBase + (lane & 15)) * K + kk + 64, 0, 3);
        #pragma unroll
        for (int t = 0; t < 4; t++) {
            acc[0][t] = WMMA_F16(a[cb][0].v, b[cb][t], acc[0][t]);
            acc[1][t] = WMMA_F16(a[cb][1].v, b[cb][t], acc[1][t]);
        }
    }
    _Float16* dst = (nBase < D_INNER) ? xin : zz;
    int nOff = (nBase < D_INNER) ? nBase : nBase - D_INNER;
    #pragma unroll
    for (int i = 0; i < 2; i++) {
        #pragma unroll
        for (int t = 0; t < 4; t++)
            store_tile_f16(acc[i][t], dst, D_INNER, mBase + i * 16, nOff + t * 16, lane);
    }
}

// ---------------------------------------------------------------------------
// Causal depthwise conv (k=4, left pad 3) + SiLU: x_in -> xc (f16)
// ---------------------------------------------------------------------------
__global__ void __launch_bounds__(256) conv_silu(const _Float16* __restrict__ xin,
                                                 const float* __restrict__ cw,
                                                 const float* __restrict__ cb,
                                                 _Float16* __restrict__ xc) {
    int idx = blockIdx.x * blockDim.x + threadIdx.x;
    if (idx >= BL * D_INNER) return;
    int e  = idx & (D_INNER - 1);
    int bl = idx >> 9;
    int l  = bl & (LL - 1);
    float acc = cb[e];
    #pragma unroll
    for (int t = 0; t < 4; t++) {
        int dl = l - 3 + t;
        if (dl >= 0)
            acc += cw[e * 4 + t] * (float)xin[(size_t)(bl - 3 + t) * D_INNER + e];
    }
    xc[idx] = (_Float16)(acc * sigmoidf_(acc));
}

// ---------------------------------------------------------------------------
// GEMM2: dbc = xc(32768x512) * x_proj_w^T(512x48)  (f32 out)
// grid 256, block 128; wave = 32x48 tile, double-buffered K
// ---------------------------------------------------------------------------
__global__ void __launch_bounds__(128) gemm_xp(const _Float16* __restrict__ A,
                                               const _Float16* __restrict__ W,
                                               float* __restrict__ dbc) {
    const int K = 512;
    int lane  = threadIdx.x & 31;
    int wv    = threadIdx.x >> 5;
    int mBase = blockIdx.x * 128 + wv * 32;
    v8f acc[2][3] = {};
    F16Frag a[2][2];
    v16h    b[2][3];
    load_a_frag(a[0][0], A, K, mBase,      0, lane);
    load_a_frag(a[0][1], A, K, mBase + 16, 0, lane);
    #pragma unroll
    for (int t = 0; t < 3; t++) b[0][t] = load_b_frag(W, K, t * 16, 0, lane);
    #pragma unroll
    for (int kk = 0; kk < K; kk += 32) {
        const int cb = (kk >> 5) & 1, nb = cb ^ 1;
        if (kk + 32 < K) {
            load_a_frag(a[nb][0], A, K, mBase,      kk + 32, lane);
            load_a_frag(a[nb][1], A, K, mBase + 16, kk + 32, lane);
            #pragma unroll
            for (int t = 0; t < 3; t++) b[nb][t] = load_b_frag(W, K, t * 16, kk + 32, lane);
        }
        if (kk + 64 < K)
            __builtin_prefetch(A + (size_t)(mBase + (lane & 15)) * K + kk + 64, 0, 3);
        #pragma unroll
        for (int t = 0; t < 3; t++) {
            acc[0][t] = WMMA_F16(a[cb][0].v, b[cb][t], acc[0][t]);
            acc[1][t] = WMMA_F16(a[cb][1].v, b[cb][t], acc[1][t]);
        }
    }
    #pragma unroll
    for (int i = 0; i < 2; i++) {
        #pragma unroll
        for (int t = 0; t < 3; t++)
            store_tile_f32(acc[i][t], dbc, 48, mBase + i * 16, t * 16, lane);
    }
}

// ---------------------------------------------------------------------------
// Selective scan: lane = one (channel, state) pair -> 131072 lanes (4096 waves).
// Fuses dt_proj + softplus + gating.  dbc rows (48 f32 per (b,l), shared by all
// channels of a batch) staged into LDS in 64-step / 12KB chunks via the TDM.
// Block: 256 threads = 16 channels of one batch; grid = 512 blocks.
// ---------------------------------------------------------------------------
__global__ void __launch_bounds__(256) scan_kernel(const float* __restrict__ dbc,
                                                   const _Float16* __restrict__ xc,
                                                   const _Float16* __restrict__ zz,
                                                   const float* __restrict__ A_log,
                                                   const float* __restrict__ dtw_,
                                                   const float* __restrict__ dtb,
                                                   const float* __restrict__ Dp_,
                                                   _Float16* __restrict__ y) {
    __shared__ float rows[64 * 48];
    int tid = threadIdx.x;
    int j   = tid & 15;                          // state index
    int ch  = tid >> 4;                          // channel slot in block (0..15)
    int b   = blockIdx.x >> 5;                   // 32 blocks per batch
    int e   = ((blockIdx.x & 31) << 4) + ch;     // channel 0..511
    float Aj   = -__expf(A_log[e * D_STATE + j]);
    float dtwj = dtw_[e * DT_RANK + j];
    float bias = dtb[e], Dp = Dp_[e];
    float h = 0.0f;
    const size_t baseBL = (size_t)b * LL;
#if USE_TDM
    unsigned lds_rows = (unsigned)(size_t)&rows[0];   // LDS aperture: low 32 bits
#endif
    for (int l0 = 0; l0 < LL; l0 += 64) {
        __syncthreads();
#if USE_TDM
        if ((tid >> 5) == 0) {                   // one TDM descriptor per chunk
            tdm_copy_12k(dbc + (baseBL + l0) * 48, lds_rows);
            __builtin_amdgcn_s_wait_tensorcnt(0);
        }
#else
        {
            const float4* src  = (const float4*)(dbc + (baseBL + l0) * 48);
            float4*       dst4 = (float4*)rows;
            for (int i = tid; i < 768; i += 256) dst4[i] = src[i];
        }
#endif
        __syncthreads();
        for (int t = 0; t < 64; t++) {
            const float* r = rows + t * 48;
            // dt = softplus( dot(dbc[:16], dt_proj_w[e]) + bias ), 16-lane reduce
            float acc = r[j] * dtwj;
            acc += __shfl_xor(acc, 1, 32);
            acc += __shfl_xor(acc, 2, 32);
            acc += __shfl_xor(acc, 4, 32);
            acc += __shfl_xor(acc, 8, 32);
            acc += bias;
            float dt = (acc > 20.0f) ? acc : log1pf(__expf(acc));
            size_t off = (baseBL + l0 + t) * D_INNER + e;
            float xt = (float)xc[off];
            float zt = (float)zz[off];
            // h_j = exp(dt*A_j)*h_j + dt*B_j*x ;  y = sum_j h_j*C_j
            h = __expf(dt * Aj) * h + (dt * r[16 + j]) * xt;
            float ys = h * r[32 + j];
            ys += __shfl_xor(ys, 1, 32);
            ys += __shfl_xor(ys, 2, 32);
            ys += __shfl_xor(ys, 4, 32);
            ys += __shfl_xor(ys, 8, 32);
            if (j == 0) {
                float yv = (ys + Dp * xt) * (zt * sigmoidf_(zt));
                y[off] = (_Float16)yv;
            }
        }
    }
}

// ---------------------------------------------------------------------------
// GEMM3: out = y(32768x512) * out_proj_w^T(512x256) + x  (f32 out, residual)
// grid (256,4), block 128; wave = 32x64 tile, double-buffered K
// ---------------------------------------------------------------------------
__global__ void __launch_bounds__(128) gemm_out(const _Float16* __restrict__ A,
                                                const _Float16* __restrict__ W,
                                                const float* __restrict__ xres,
                                                float* __restrict__ out) {
    const int K = 512;
    int lane  = threadIdx.x & 31;
    int wv    = threadIdx.x >> 5;
    int mBase = blockIdx.x * 128 + wv * 32;
    int nBase = blockIdx.y * 64;
    v8f acc[2][4] = {};
    F16Frag a[2][2];
    v16h    b[2][4];
    load_a_frag(a[0][0], A, K, mBase,      0, lane);
    load_a_frag(a[0][1], A, K, mBase + 16, 0, lane);
    #pragma unroll
    for (int t = 0; t < 4; t++) b[0][t] = load_b_frag(W, K, nBase + t * 16, 0, lane);
    #pragma unroll
    for (int kk = 0; kk < K; kk += 32) {
        const int cb = (kk >> 5) & 1, nb = cb ^ 1;
        if (kk + 32 < K) {
            load_a_frag(a[nb][0], A, K, mBase,      kk + 32, lane);
            load_a_frag(a[nb][1], A, K, mBase + 16, kk + 32, lane);
            #pragma unroll
            for (int t = 0; t < 4; t++) b[nb][t] = load_b_frag(W, K, nBase + t * 16, kk + 32, lane);
        }
        if (kk + 64 < K)
            __builtin_prefetch(A + (size_t)(mBase + (lane & 15)) * K + kk + 64, 0, 3);
        #pragma unroll
        for (int t = 0; t < 4; t++) {
            acc[0][t] = WMMA_F16(a[cb][0].v, b[cb][t], acc[0][t]);
            acc[1][t] = WMMA_F16(a[cb][1].v, b[cb][t], acc[1][t]);
        }
    }
    int nc = lane & 15;
    #pragma unroll
    for (int i = 0; i < 2; i++) {
        int m0 = mBase + i * 16 + ((lane >> 4) << 3);
        #pragma unroll
        for (int t = 0; t < 4; t++) {
            int n = nBase + t * 16 + nc;
            #pragma unroll
            for (int r = 0; r < 8; r++) {
                size_t o = (size_t)(m0 + r) * D_MODEL + n;
                out[o] = acc[i][t][r] + xres[o];
            }
        }
    }
}

// ---------------------------------------------------------------------------
extern "C" void kernel_launch(void* const* d_in, const int* in_sizes, int n_in,
                              void* d_out, int out_size, void* d_ws, size_t ws_size,
                              hipStream_t stream) {
    const float* x         = (const float*)d_in[0];
    const float* ln_w      = (const float*)d_in[1];
    const float* ln_b      = (const float*)d_in[2];
    const float* in_proj_w = (const float*)d_in[3];
    const float* conv_w    = (const float*)d_in[4];
    const float* conv_b    = (const float*)d_in[5];
    const float* x_proj_w  = (const float*)d_in[6];
    const float* dt_proj_w = (const float*)d_in[7];
    const float* dt_proj_b = (const float*)d_in[8];
    const float* A_log     = (const float*)d_in[9];
    const float* D_param   = (const float*)d_in[10];
    const float* out_proj_w= (const float*)d_in[11];

    char* ws = (char*)d_ws;
    size_t o = 0;
    auto alloc = [&](size_t bytes) -> void* {
        void* p = ws + o;
        o += (bytes + 255) & ~(size_t)255;
        return p;
    };
    _Float16* xn_h    = (_Float16*)alloc((size_t)BL * D_MODEL * 2);        // 16.8 MB
    _Float16* w_in_h  = (_Float16*)alloc((size_t)2*D_INNER * D_MODEL * 2); // 0.5 MB
    _Float16* w_xp_h  = (_Float16*)alloc((size_t)48 * D_INNER * 2);
    _Float16* w_out_h = (_Float16*)alloc((size_t)D_MODEL * D_INNER * 2);
    _Float16* xin_h   = (_Float16*)alloc((size_t)BL * D_INNER * 2);        // 33.5 MB
    _Float16* z_h     = (_Float16*)alloc((size_t)BL * D_INNER * 2);        // 33.5 MB
    _Float16* xc_h    = (_Float16*)alloc((size_t)BL * D_INNER * 2);        // 33.5 MB
    // buffer reuse: dbc overwrites dead xn_h; y overwrites dead xin_h
    float*    dbc     = (float*)xn_h;     // 32768*48*4 = 6.3 MB <= 16.8 MB
    _Float16* y_h     = xin_h;            // same size, xin dead after conv

    ln_kernel<<<BL / 8, 256, 0, stream>>>(x, ln_w, ln_b, xn_h);
    cvt_f32_f16<<<(2*D_INNER*D_MODEL + 255) / 256, 256, 0, stream>>>(in_proj_w, w_in_h, 2*D_INNER*D_MODEL);
    cvt_f32_f16<<<(48*D_INNER + 255) / 256, 256, 0, stream>>>(x_proj_w, w_xp_h, 48*D_INNER);
    cvt_f32_f16<<<(D_MODEL*D_INNER + 255) / 256, 256, 0, stream>>>(out_proj_w, w_out_h, D_MODEL*D_INNER);

    gemm_in<<<dim3(BL / 128, (2*D_INNER) / 64), 128, 0, stream>>>(xn_h, w_in_h, xin_h, z_h);
    conv_silu<<<(BL * D_INNER) / 256, 256, 0, stream>>>(xin_h, conv_w, conv_b, xc_h);
    gemm_xp<<<BL / 128, 128, 0, stream>>>(xc_h, w_xp_h, dbc);
    scan_kernel<<<NB * 32, 256, 0, stream>>>(dbc, xc_h, z_h, A_log, dt_proj_w, dt_proj_b, D_param, y_h);
    gemm_out<<<dim3(BL / 128, D_MODEL / 64), 128, 0, stream>>>(y_h, w_out_h, x, (float*)d_out);
}
